// CausalSelfAttention_77953656422509
// MI455X (gfx1250) — matrix operand order repaired and checked by reading, not compiled, not run on hardware
//
#include <hip/hip_runtime.h>

// ---------------- types ----------------
typedef unsigned int  u32;
typedef unsigned short u16;
typedef __attribute__((ext_vector_type(2)))  u32   u32x2;
typedef __attribute__((ext_vector_type(4)))  u32   u32x4;
typedef __attribute__((ext_vector_type(4)))  int   i32x4;
typedef __attribute__((ext_vector_type(4)))  float f32x4;
typedef __attribute__((ext_vector_type(16))) __bf16 v16bf;
typedef __attribute__((ext_vector_type(8)))  float v8f;

union Frag { u32x4 u[2]; v16bf v; };

#define BATCH 4
#define SEQ   2048
#define DIM   1024
#define MTOT  (BATCH*SEQ)   // 8192
#define LDSTR 40            // LDS row stride in bf16 elems (80B, multiple of 16B)

// ---------------- CDNA5 async global->LDS path (guarded) ----------------
#if defined(__has_builtin)
#if __has_builtin(__builtin_amdgcn_global_load_async_to_lds_b128) && \
    __has_builtin(__builtin_amdgcn_s_wait_asynccnt)
#define HAVE_ASYNC 1
#endif
#endif

#define GLOBAL_AS __attribute__((address_space(1)))
#define LDS_AS    __attribute__((address_space(3)))

// 16-byte copy global -> LDS. Async (ASYNCcnt-tracked, bypasses VGPRs) when
// the toolchain exposes the gfx1250 builtin; synchronous fallback otherwise.
__device__ __forceinline__ void cp16_g2l(const u16* g, u16* l)
{
#ifdef HAVE_ASYNC
    __builtin_amdgcn_global_load_async_to_lds_b128(
        (GLOBAL_AS i32x4*)g, (LDS_AS i32x4*)l, 0, 0);
#else
    *(u32x4*)l = *(const u32x4*)g;
#endif
}

#ifdef HAVE_ASYNC
#define ASYNC_WAIT(n) __builtin_amdgcn_s_wait_asynccnt(n)
#else
#define ASYNC_WAIT(n) ((void)0)
#endif

// ---------------- helpers ----------------
__device__ __forceinline__ u32 f32_bits(float f) { union { float f; u32 u; } c; c.f = f; return c.u; }
__device__ __forceinline__ u16 f2bf(float f)     { return (u16)(f32_bits(f) >> 16); }
__device__ __forceinline__ float bf2f(u16 h)     { union { u32 u; float f; } c; c.u = ((u32)h) << 16; return c.f; }
__device__ __forceinline__ u32 pack2(float a, float b) { return (f32_bits(b) & 0xFFFF0000u) | (f32_bits(a) >> 16); }

// One K-step of the 128x128 workgroup tile: each of 8 waves does a 64x32 tile
// (4x2 WMMA 16x16x32_bf16). A is row-major [128][LDSTR] bf16 in LDS,
// B is staged transposed as Bt[N=128][K=32] (stride LDSTR) in LDS.
__device__ __forceinline__ void wave_mma_step(const u16* As, const u16* Bs,
                                              int wm, int wn, int lane,
                                              v8f acc[4][2])
{
    const int r15 = lane & 15;
    const int kh  = lane >> 4;
    Frag a[4], b[2];
#pragma unroll
    for (int mi = 0; mi < 4; ++mi) {
        const u32x4* p = (const u32x4*)(As + (wm * 64 + mi * 16 + r15) * LDSTR);
        a[mi].u[0] = p[kh];       // K = kh*8 + {0,2,4,6} pairs
        a[mi].u[1] = p[2 + kh];   // K = 16 + kh*8 + ...
    }
#pragma unroll
    for (int ni = 0; ni < 2; ++ni) {
        const u32x4* p = (const u32x4*)(Bs + (wn * 32 + ni * 16 + r15) * LDSTR);
        b[ni].u[0] = p[kh];       // K = kh*8 .. kh*8+7 for column r15
        b[ni].u[1] = p[2 + kh];   // K = 16 + kh*8 + ...
    }
#pragma unroll
    for (int mi = 0; mi < 4; ++mi)
#pragma unroll
        for (int ni = 0; ni < 2; ++ni)
            acc[mi][ni] = __builtin_amdgcn_wmma_f32_16x16x32_bf16(
                false, a[mi].v, false, b[ni].v, (short)0, acc[mi][ni], false, false);
}

__device__ __forceinline__ void zero_acc(v8f acc[4][2])
{
#pragma unroll
    for (int mi = 0; mi < 4; ++mi)
#pragma unroll
        for (int ni = 0; ni < 2; ++ni)
#pragma unroll
            for (int t = 0; t < 8; ++t) acc[mi][ni][t] = 0.0f;
}

// Stage a 128x32 bf16 slab (row-major source) into LDS rows. 2 x b128 / thread.
__device__ __forceinline__ void stage_rows(const u16* src, int rowBase, int srcStride,
                                           int k0, u16* dst, int tid)
{
#pragma unroll
    for (int i = 0; i < 2; ++i) {
        int idx = tid + 256 * i;           // 512 x 16B
        int row = idx >> 2;
        int h   = (idx & 3) * 8;
        cp16_g2l(src + (size_t)(rowBase + row) * srcStride + k0 + h,
                 dst + row * LDSTR + h);
    }
}

// ---------------- kernel 1: fused QKV projection ----------------
// q/k = x @ W + b, bf16 [B*S][D]. v is written TRANSPOSED as [B][D][S] so the
// PV GEMM can stage it with straight (async) row copies.
// grid (N/128=8, M/128=64, 3)
__global__ __launch_bounds__(256) void proj_kernel(
    const float* __restrict__ x,
    const float* __restrict__ Wq, const float* __restrict__ bq,
    const float* __restrict__ Wk, const float* __restrict__ bk,
    const float* __restrict__ Wv, const float* __restrict__ bv,
    u16* __restrict__ qb, u16* __restrict__ kb, u16* __restrict__ vbT)
{
    __shared__ __align__(16) u16 As[128 * LDSTR];
    __shared__ __align__(16) u16 Bs[128 * LDSTR];

    const int tid  = threadIdx.x;
    const int lane = tid & 31;
    const int wave = tid >> 5;
    const int wm = wave & 1, wn = wave >> 1;
    const int nBase = blockIdx.x * 128;
    const int mBase = blockIdx.y * 128;
    const int z = blockIdx.z;
    const float* W    = (z == 0) ? Wq : (z == 1) ? Wk : Wv;
    const float* bias = (z == 0) ? bq : (z == 1) ? bk : bv;

    v8f acc[4][2];
    zero_acc(acc);

    for (int k0 = 0; k0 < DIM; k0 += 32) {
        // stage A: 128x32 f32 -> bf16, row major (conversion => through VGPRs)
#pragma unroll
        for (int i = 0; i < 4; ++i) {
            int idx = tid + 256 * i;                // 1024 float4 slots
            int row = idx >> 3;
            int c4  = (idx & 7) * 4;
            f32x4 v = *(const f32x4*)(x + (size_t)(mBase + row) * DIM + k0 + c4);
            u32x2 pk; pk.x = pack2(v.x, v.y); pk.y = pack2(v.z, v.w);
            *(u32x2*)(As + row * LDSTR + c4) = pk;
        }
        // stage B transposed: Bt[n][k] = W[k0+k][nBase+n]
#pragma unroll
        for (int i = 0; i < 4; ++i) {
            int idx = tid + 256 * i;                // 1024 float4 slots
            int k = idx >> 5;
            int n = (idx & 31) * 4;
            f32x4 v = *(const f32x4*)(W + (size_t)(k0 + k) * DIM + nBase + n);
            Bs[(n + 0) * LDSTR + k] = f2bf(v.x);
            Bs[(n + 1) * LDSTR + k] = f2bf(v.y);
            Bs[(n + 2) * LDSTR + k] = f2bf(v.z);
            Bs[(n + 3) * LDSTR + k] = f2bf(v.w);
        }
        __syncthreads();
        wave_mma_step(As, Bs, wm, wn, lane, acc);
        __syncthreads();
    }

    const int r15 = lane & 15, kh = lane >> 4;
    if (z < 2) {
        u16* out = (z == 0) ? qb : kb;
#pragma unroll
        for (int mi = 0; mi < 4; ++mi)
#pragma unroll
            for (int ni = 0; ni < 2; ++ni) {
                int gn = nBase + wn * 32 + ni * 16 + r15;
                float bb = bias[gn];
#pragma unroll
                for (int r = 0; r < 8; ++r) {
                    int gm = mBase + wm * 64 + mi * 16 + r + 8 * kh;
                    out[(size_t)gm * DIM + gn] = f2bf(acc[mi][ni][r] + bb);
                }
            }
    } else {
        // V: store transposed [B][D][S]; the 128-row tile is within one batch.
        const int b     = mBase >> 11;          // mBase / SEQ
        const int sBase = mBase & (SEQ - 1);
#pragma unroll
        for (int mi = 0; mi < 4; ++mi)
#pragma unroll
            for (int ni = 0; ni < 2; ++ni) {
                int gn = nBase + wn * 32 + ni * 16 + r15;
                float bb = bias[gn];
#pragma unroll
                for (int r = 0; r < 8; ++r) {
                    int s = sBase + wm * 64 + mi * 16 + r + 8 * kh;
                    vbT[((size_t)b * DIM + gn) * SEQ + s] = f2bf(acc[mi][ni][r] + bb);
                }
            }
    }
}

// ---------------- kernel 2: scores = Q K^T with causal mask ----------------
// A = q bf16 [S][D]; Bt(direct) = k bf16 [S][D]; out = sc bf16 [S][S]
// Async double-buffered staging. grid (S/128=16, S/128=16, B=4)
__global__ __launch_bounds__(256) void scores_kernel(
    const u16* __restrict__ qb, const u16* __restrict__ kb, u16* __restrict__ sc)
{
    __shared__ __align__(16) u16 As[2][128 * LDSTR];
    __shared__ __align__(16) u16 Bs[2][128 * LDSTR];

    const int tid  = threadIdx.x;
    const int lane = tid & 31;
    const int wave = tid >> 5;
    const int wm = wave & 1, wn = wave >> 1;
    const int kBase = blockIdx.x * 128;
    const int qBase = blockIdx.y * 128;
    const int batch = blockIdx.z;
    const u16 negb = f2bf(-1.0e9f);

    u16* scb = sc + (size_t)batch * SEQ * SEQ;

    if (kBase > qBase + 127) {   // fully above the diagonal: splat -NEG, no GEMM
        u32 pp = (u32)negb | ((u32)negb << 16);
        u32x4 fill; fill.x = pp; fill.y = pp; fill.z = pp; fill.w = pp;
#pragma unroll
        for (int i = 0; i < 8; ++i) {
            int idx = tid + 256 * i;            // 2048 x uint4 (8 bf16)
            int row = idx >> 4;
            int col = (idx & 15) * 8;
            *(u32x4*)(scb + (size_t)(qBase + row) * SEQ + kBase + col) = fill;
        }
        return;
    }

    const u16* A  = qb + (size_t)batch * SEQ * DIM;
    const u16* Bt = kb + (size_t)batch * SEQ * DIM;   // K^T in N-major == K row-major

    v8f acc[4][2];
    zero_acc(acc);

    stage_rows(A,  qBase, DIM, 0, As[0], tid);
    stage_rows(Bt, kBase, DIM, 0, Bs[0], tid);

    int cur = 0;
    for (int k0 = 0; k0 < DIM; k0 += 32) {
        if (k0 + 32 < DIM) {
            stage_rows(A,  qBase, DIM, k0 + 32, As[cur ^ 1], tid);
            stage_rows(Bt, kBase, DIM, k0 + 32, Bs[cur ^ 1], tid);
            ASYNC_WAIT(4);       // only the just-issued batch may remain in flight
        } else {
            ASYNC_WAIT(0);
        }
        __syncthreads();
        wave_mma_step(As[cur], Bs[cur], wm, wn, lane, acc);
        __syncthreads();
        cur ^= 1;
    }

    const int r15 = lane & 15, kh = lane >> 4;
#pragma unroll
    for (int mi = 0; mi < 4; ++mi)
#pragma unroll
        for (int ni = 0; ni < 2; ++ni) {
            int gk = kBase + wn * 32 + ni * 16 + r15;
#pragma unroll
            for (int r = 0; r < 8; ++r) {
                int gq = qBase + wm * 64 + mi * 16 + r + 8 * kh;
                u16 o = (gk <= gq) ? f2bf(acc[mi][ni][r]) : negb;
                scb[(size_t)gq * SEQ + gk] = o;
            }
        }
}

// ---------------- kernel 3: row softmax (in place, bf16) ----------------
// grid (S, B), 256 threads, 8 elems/thread
__global__ __launch_bounds__(256) void softmax_kernel(u16* __restrict__ sc)
{
    const int row   = blockIdx.x;
    const int batch = blockIdx.y;
    u16* p = sc + ((size_t)batch * SEQ + row) * SEQ;
    __shared__ float red[256];
    const int tid = threadIdx.x;

    float v[8];
    float m = -3.0e38f;
#pragma unroll
    for (int i = 0; i < 8; ++i) { v[i] = bf2f(p[tid + 256 * i]); m = fmaxf(m, v[i]); }
    red[tid] = m; __syncthreads();
    for (int s = 128; s > 0; s >>= 1) {
        if (tid < s) red[tid] = fmaxf(red[tid], red[tid + s]);
        __syncthreads();
    }
    m = red[0]; __syncthreads();

    float sum = 0.0f;
#pragma unroll
    for (int i = 0; i < 8; ++i) { v[i] = __expf(v[i] - m); sum += v[i]; }
    red[tid] = sum; __syncthreads();
    for (int s = 128; s > 0; s >>= 1) {
        if (tid < s) red[tid] += red[tid + s];
        __syncthreads();
    }
    float inv = 1.0f / red[0];
#pragma unroll
    for (int i = 0; i < 8; ++i) p[tid + 256 * i] = f2bf(v[i] * inv);
}

// ---------------- kernel 4: O = P V ----------------
// A = probs bf16 [S][S] (K=2048); Bt(direct) = vT bf16 [D][S]; out f32.
// Async double-buffered staging. grid (D/128=8, S/128=16, B=4)
__global__ __launch_bounds__(256) void pv_kernel(
    const u16* __restrict__ sc, const u16* __restrict__ vbT, float* __restrict__ out)
{
    __shared__ __align__(16) u16 As[2][128 * LDSTR];
    __shared__ __align__(16) u16 Bs[2][128 * LDSTR];

    const int tid  = threadIdx.x;
    const int lane = tid & 31;
    const int wave = tid >> 5;
    const int wm = wave & 1, wn = wave >> 1;
    const int nBase = blockIdx.x * 128;     // over D
    const int mBase = blockIdx.y * 128;     // over S
    const int batch = blockIdx.z;

    const u16* A  = sc  + (size_t)batch * SEQ * SEQ;   // [S][S]
    const u16* Bt = vbT + (size_t)batch * DIM * SEQ;   // [D][S] == V^T rows

    v8f acc[4][2];
    zero_acc(acc);

    stage_rows(A,  mBase, SEQ, 0, As[0], tid);
    stage_rows(Bt, nBase, SEQ, 0, Bs[0], tid);

    int cur = 0;
    for (int k0 = 0; k0 < SEQ; k0 += 32) {
        if (k0 + 32 < SEQ) {
            stage_rows(A,  mBase, SEQ, k0 + 32, As[cur ^ 1], tid);
            stage_rows(Bt, nBase, SEQ, k0 + 32, Bs[cur ^ 1], tid);
            ASYNC_WAIT(4);
        } else {
            ASYNC_WAIT(0);
        }
        __syncthreads();
        wave_mma_step(As[cur], Bs[cur], wm, wn, lane, acc);
        __syncthreads();
        cur ^= 1;
    }

    const int r15 = lane & 15, kh = lane >> 4;
    float* ob = out + (size_t)batch * SEQ * DIM;
#pragma unroll
    for (int mi = 0; mi < 4; ++mi)
#pragma unroll
        for (int ni = 0; ni < 2; ++ni) {
            int gn = nBase + wn * 32 + ni * 16 + r15;
#pragma unroll
            for (int r = 0; r < 8; ++r) {
                int gm = mBase + wm * 64 + mi * 16 + r + 8 * kh;
                ob[(size_t)gm * DIM + gn] = acc[mi][ni][r];
            }
        }
}

// ---------------- launcher ----------------
extern "C" void kernel_launch(void* const* d_in, const int* in_sizes, int n_in,
                              void* d_out, int out_size, void* d_ws, size_t ws_size,
                              hipStream_t stream)
{
    const float* x  = (const float*)d_in[0];
    const float* Wq = (const float*)d_in[1];
    const float* bq = (const float*)d_in[2];
    const float* Wk = (const float*)d_in[3];
    const float* bk = (const float*)d_in[4];
    const float* Wv = (const float*)d_in[5];
    const float* bv = (const float*)d_in[6];
    float* out = (float*)d_out;

    // workspace layout (bf16):
    //   qb [8192*1024] | kb [8192*1024] | vbT [4*1024*2048] | sc [4*2048*2048]
    u16* qb  = (u16*)d_ws;
    u16* kb  = qb  + (size_t)MTOT * DIM;
    u16* vbT = kb  + (size_t)MTOT * DIM;
    u16* sc  = vbT + (size_t)BATCH * DIM * SEQ;

    dim3 blk(256, 1, 1);

    dim3 gProj(DIM / 128, MTOT / 128, 3);
    hipLaunchKernelGGL(proj_kernel, gProj, blk, 0, stream,
                       x, Wq, bq, Wk, bk, Wv, bv, qb, kb, vbT);

    dim3 gScore(SEQ / 128, SEQ / 128, BATCH);
    hipLaunchKernelGGL(scores_kernel, gScore, blk, 0, stream, qb, kb, sc);

    dim3 gSoft(SEQ, BATCH, 1);
    hipLaunchKernelGGL(softmax_kernel, gSoft, blk, 0, stream, sc);

    dim3 gPV(DIM / 128, SEQ / 128, BATCH);
    hipLaunchKernelGGL(pv_kernel, gPV, blk, 0, stream, sc, vbT, out);
}
